// DualPathRecurrentProcessor_79577154060725
// MI455X (gfx1250) — matrix operand order repaired and checked by reading, not compiled, not run on hardware
//
#include <hip/hip_runtime.h>
#include <hip/hip_bf16.h>
#include <math.h>

// CDNA5 WMMA vector types (probe-confirmed signatures)
typedef __attribute__((ext_vector_type(16))) __bf16 v16bf;
typedef __attribute__((ext_vector_type(8)))  float  v8f;

#define DIM      128
#define EXPN     256            // internal width per path
#define SEQ      2048
#define BATCH    32
#define M_TOTAL  (BATCH * SEQ)  // 65536
#define LOG3F    1.0986122886681098f
#define REC_COEF (3.0f / 2047.0f)

__device__ __forceinline__ float sigmoid_f(float x) {
  return 1.0f / (1.0f + __expf(-x));
}

// 32-byte WMMA fragment <- two global_load_b128
union V16 { v16bf bf; uint4 u4[2]; };

// A-fragment: lane element e holds k = (e&7) + 16*(e>>3)  -> runs [0..7],[16..23]
__device__ __forceinline__ v16bf load_frag_a(const __bf16* __restrict__ p) {
  V16 t;
  t.u4[0] = *(const uint4*)(p);
  t.u4[1] = *(const uint4*)(p + 16);
  return t.bf;
}
// B-fragment: lane element e holds k = (e&7) + 16*(e>>3)  -> runs [0..7],[16..23]
__device__ __forceinline__ v16bf load_frag_b(const __bf16* __restrict__ p) {
  V16 t;
  t.u4[0] = *(const uint4*)(p);
  t.u4[1] = *(const uint4*)(p + 16);
  return t.bf;
}

#define WMMA_BF16(A, B, C) \
  __builtin_amdgcn_wmma_f32_16x16x32_bf16(false, (A), false, (B), (short)0, (C), false, false)

// ---------------------------------------------------------------------------
// Prep 1: split x into bf16 hi/lo planes, plain (path 0) and recency-weighted
// (path 1; weight is a pure row scale so it folds into the operand).
// ---------------------------------------------------------------------------
__global__ __launch_bounds__(256)
void prep_x_kernel(const float* __restrict__ x,
                   __bf16* __restrict__ h0, __bf16* __restrict__ l0,
                   __bf16* __restrict__ h1, __bf16* __restrict__ l1) {
  size_t i = (size_t)blockIdx.x * blockDim.x + threadIdx.x;  // < 65536*128
  int m = (int)(i >> 7);
  float v = x[i];
  __bf16 hb = (__bf16)v;
  h0[i] = hb;
  l0[i] = (__bf16)(v - (float)hb);
  float vw = v * __expf((float)(m & (SEQ - 1)) * REC_COEF);
  __bf16 hw = (__bf16)vw;
  h1[i] = hw;
  l1[i] = (__bf16)(vw - (float)hw);
}

// Prep 2: generic f32 -> bf16 hi/lo splitter (for the weight matrices).
__global__ __launch_bounds__(256)
void split_kernel(const float* __restrict__ src,
                  __bf16* __restrict__ hi, __bf16* __restrict__ lo, int n) {
  int i = blockIdx.x * blockDim.x + threadIdx.x;
  if (i < n) {
    float v = src[i];
    __bf16 h = (__bf16)v;
    hi[i] = h;
    lo[i] = (__bf16)(v - (float)h);
  }
}

// ---------------------------------------------------------------------------
// Fused projection + gate math + sequential scan (one path per launch).
// Block = 128 threads = 4 waves; wave owns a 16-column tile; block covers 64
// of 256 channels for one batch row. Weights live in registers for the whole
// sequence (16 v16bf frags). Loop over 128 s-tiles of 16 steps:
//   WMMA (split-precision bf16, 24 v_wmma) -> gate math -> in-register scan.
// The 16-step scan uses the affine trick h_out = P*h_in + S: each half-wave
// scans its 8 rows in parallel, two shuffles stitch halves / carry h.
// ---------------------------------------------------------------------------
__global__ __launch_bounds__(128)
void fused_proj_scan_kernel(const __bf16* __restrict__ x_hi, const __bf16* __restrict__ x_lo,
                            const __bf16* __restrict__ wa_hi, const __bf16* __restrict__ wa_lo,
                            const __bf16* __restrict__ wi_hi, const __bf16* __restrict__ wi_lo,
                            const float* __restrict__ ba, const float* __restrict__ bi,
                            const float* __restrict__ g,
                            __bf16* __restrict__ comb_hi, __bf16* __restrict__ comb_lo,
                            int path) {
  const int lane    = threadIdx.x & 31;
  const int wave    = threadIdx.x >> 5;
  const int r       = lane & 15;
  const int hi_half = lane >> 4;
  const int ka      = hi_half << 3;    // A-frag K base inside 32-chunk
  const int kb      = hi_half << 3;    // B-frag K base inside 32-chunk

  const int b  = blockIdx.y;
  const int n0 = blockIdx.x * 64 + wave * 16;
  const int n  = n0 + r;               // this lane's column (weights row)

  // Resident B fragments for Wa and Wi, all 4 K-chunks, hi/lo planes.
  v16bf BAhi[4], BAlo[4], BIhi[4], BIlo[4];
#pragma unroll
  for (int c = 0; c < 4; ++c) {
    const size_t wo = (size_t)n * DIM + c * 32 + kb;
    BAhi[c] = load_frag_b(wa_hi + wo);
    BAlo[c] = load_frag_b(wa_lo + wo);
    BIhi[c] = load_frag_b(wi_hi + wo);
    BIlo[c] = load_frag_b(wi_lo + wo);
  }
  const float bav   = ba[n];
  const float biv   = bi[n];
  const float alpha = sigmoid_f(g[n]);

  const int lo_partner = lane & 15;          // low-half lane, same column
  const int hi_partner = (lane & 15) + 16;   // high-half lane, same column

  float h_carry = 0.0f;
  const size_t cbase = ((size_t)path * M_TOTAL + (size_t)b * SEQ) * EXPN + n;

  for (int st = 0; st < SEQ / 16; ++st) {
    const size_t arow = ((size_t)b * SEQ + st * 16 + r) * DIM;
    if (st + 1 < SEQ / 16) {  // pull next tile toward the WGP (global_prefetch_b8)
      __builtin_prefetch(x_hi + arow + 16 * DIM, 0, 0);
      __builtin_prefetch(x_lo + arow + 16 * DIM, 0, 0);
    }

    v8f acc_a = {};
    v8f acc_i = {};
#pragma unroll
    for (int c = 0; c < 4; ++c) {
      v16bf Ahi = load_frag_a(x_hi + arow + c * 32 + ka);
      v16bf Alo = load_frag_a(x_lo + arow + c * 32 + ka);
      acc_a = WMMA_BF16(Ahi, BAlo[c], acc_a);
      acc_a = WMMA_BF16(Alo, BAhi[c], acc_a);
      acc_a = WMMA_BF16(Ahi, BAhi[c], acc_a);
      acc_i = WMMA_BF16(Ahi, BIlo[c], acc_i);
      acc_i = WMMA_BF16(Alo, BIhi[c], acc_i);
      acc_i = WMMA_BF16(Ahi, BIhi[c], acc_i);
    }

    // Gate math. Lane element rr covers step s = st*16 + rr + 8*hi_half.
    float av[8], uv[8];
#pragma unroll
    for (int rr = 0; rr < 8; ++rr) {
      float pa = acc_a[rr] + bav;
      float pi = acc_i[rr] + biv;
      float rg = sigmoid_f(pa);
      float ig = sigmoid_f(pi);
      float a  = alpha * __expf(-rg * LOG3F);       // alpha / 3^rg
      av[rr] = a;
      uv[rr] = sqrtf(fmaxf(1.0f - a * a, 0.0f)) * ig * pi;
    }

    // Local affine scan of this lane's 8 steps: h_j = p[j]*h_in + s[j].
    float p[8], s[8];
    p[0] = av[0];
    s[0] = uv[0];
#pragma unroll
    for (int j = 1; j < 8; ++j) {
      p[j] = p[j - 1] * av[j];
      s[j] = fmaf(av[j], s[j - 1], uv[j]);
    }
    // Stitch halves: high half starts from low half's output.
    float Plow = __shfl(p[7], lo_partner, 32);
    float Slow = __shfl(s[7], lo_partner, 32);
    float h_in = hi_half ? fmaf(Plow, h_carry, Slow) : h_carry;
    float hj[8];
#pragma unroll
    for (int j = 0; j < 8; ++j) hj[j] = fmaf(p[j], h_in, s[j]);
    // Tile output (step 15) lives in the high partner's element 7.
    h_carry = __shfl(hj[7], hi_partner, 32);

    // Emit combined activation as bf16 hi/lo planes (ready for the out-GEMM).
    size_t cidx = cbase + (size_t)(st * 16 + (hi_half << 3)) * EXPN;
#pragma unroll
    for (int j = 0; j < 8; ++j) {
      float hv = hj[j];
      __bf16 hb = (__bf16)hv;
      comb_hi[cidx] = hb;
      comb_lo[cidx] = (__bf16)(hv - (float)hb);
      cidx += EXPN;
    }
  }
}

// ---------------------------------------------------------------------------
// Output GEMM: out = combined[65536x512] . Wf^T + bf  (Wf:[128,512]).
// All operands pre-split bf16 -> pure b128 loads + 48 v_wmma per wave.
// combined feature f = path*256 + channel lives at
//   plane[(f>>8)*M_TOTAL*EXPN + m*EXPN + (f&255)].
// ---------------------------------------------------------------------------
__global__ __launch_bounds__(128)
void out_gemm_kernel(const __bf16* __restrict__ comb_hi, const __bf16* __restrict__ comb_lo,
                     const __bf16* __restrict__ wf_hi, const __bf16* __restrict__ wf_lo,
                     const float* __restrict__ bf, float* __restrict__ out) {
  const int lane    = threadIdx.x & 31;
  const int wave    = threadIdx.x >> 5;
  const int r       = lane & 15;
  const int hi_half = lane >> 4;
  const int ka      = hi_half << 3;
  const int kb      = hi_half << 3;

  const int m0 = blockIdx.x * 16;
  const int n0 = blockIdx.y * 64 + wave * 16;
  const int mA = m0 + r;
  const int nB = n0 + r;

  v8f acc = {};

#pragma unroll
  for (int kc = 0; kc < 4 * DIM; kc += 32) {
    const int f0 = kc + ka;  // runs [f0..f0+7],[f0+16..f0+23] stay in one path block
    const size_t abase = ((size_t)(f0 >> 8) * M_TOTAL + mA) * EXPN + (f0 & (EXPN - 1));
    V16 th, tl;
    th.u4[0] = *(const uint4*)(comb_hi + abase);
    th.u4[1] = *(const uint4*)(comb_hi + abase + 16);
    tl.u4[0] = *(const uint4*)(comb_lo + abase);
    tl.u4[1] = *(const uint4*)(comb_lo + abase + 16);

    const size_t wo = (size_t)nB * (4 * DIM) + kc + kb;
    v16bf Bhi = load_frag_b(wf_hi + wo);
    v16bf Blo = load_frag_b(wf_lo + wo);

    acc = WMMA_BF16(th.bf, Blo, acc);
    acc = WMMA_BF16(tl.bf, Bhi, acc);
    acc = WMMA_BF16(th.bf, Bhi, acc);
  }

  const int n = n0 + (lane & 15);
  const float bias = bf[n];
#pragma unroll
  for (int rr = 0; rr < 8; ++rr) {
    int m = m0 + rr + (hi_half << 3);
    out[(size_t)m * DIM + n] = acc[rr] + bias;
  }
}

// ---------------------------------------------------------------------------
// Launch. Workspace (all bf16 planes, ~202 MB):
//   x planes   : 4 x [65536][128]          (plain hi/lo, weighted hi/lo)
//   gate W     : 8 x [256][128]            (Wa/Wi x 2 paths x hi/lo)
//   Wf planes  : 2 x [128][512]
//   comb planes: 2 x [2*65536][256]
// ---------------------------------------------------------------------------
extern "C" void kernel_launch(void* const* d_in, const int* in_sizes, int n_in,
                              void* d_out, int out_size, void* d_ws, size_t ws_size,
                              hipStream_t stream) {
  const float* x    = (const float*)d_in[0];
  const float* Wa_u = (const float*)d_in[1];
  const float* ba_u = (const float*)d_in[2];
  const float* Wi_u = (const float*)d_in[3];
  const float* bi_u = (const float*)d_in[4];
  const float* g_u  = (const float*)d_in[5];
  const float* Wa_w = (const float*)d_in[6];
  const float* ba_w = (const float*)d_in[7];
  const float* Wi_w = (const float*)d_in[8];
  const float* bi_w = (const float*)d_in[9];
  const float* g_w  = (const float*)d_in[10];
  const float* Wf   = (const float*)d_in[11];
  const float* bfv  = (const float*)d_in[12];
  float* out = (float*)d_out;

  const size_t XN  = (size_t)M_TOTAL * DIM;   // 8388608
  const int    WGN = EXPN * DIM;              // 32768
  const int    WFN = DIM * 4 * DIM;           // 65536
  const size_t CN  = (size_t)2 * M_TOTAL * EXPN;

  char* wsp = (char*)d_ws;
  auto alloc = [&](size_t bytes) {
    char* p = wsp;
    wsp += (bytes + 255) & ~(size_t)255;
    return p;
  };
  __bf16* xh0 = (__bf16*)alloc(XN * 2);
  __bf16* xl0 = (__bf16*)alloc(XN * 2);
  __bf16* xh1 = (__bf16*)alloc(XN * 2);
  __bf16* xl1 = (__bf16*)alloc(XN * 2);
  __bf16* wauh = (__bf16*)alloc(WGN * 2);  __bf16* waul = (__bf16*)alloc(WGN * 2);
  __bf16* wiuh = (__bf16*)alloc(WGN * 2);  __bf16* wiul = (__bf16*)alloc(WGN * 2);
  __bf16* wawh = (__bf16*)alloc(WGN * 2);  __bf16* wawl = (__bf16*)alloc(WGN * 2);
  __bf16* wiwh = (__bf16*)alloc(WGN * 2);  __bf16* wiwl = (__bf16*)alloc(WGN * 2);
  __bf16* wfh  = (__bf16*)alloc(WFN * 2);  __bf16* wfl  = (__bf16*)alloc(WFN * 2);
  __bf16* ch   = (__bf16*)alloc(CN * 2);
  __bf16* cl   = (__bf16*)alloc(CN * 2);

  // --- prep: bf16 hi/lo planes
  prep_x_kernel<<<(unsigned)(XN / 256), 256, 0, stream>>>(x, xh0, xl0, xh1, xl1);
  split_kernel<<<(WGN + 255) / 256, 256, 0, stream>>>(Wa_u, wauh, waul, WGN);
  split_kernel<<<(WGN + 255) / 256, 256, 0, stream>>>(Wi_u, wiuh, wiul, WGN);
  split_kernel<<<(WGN + 255) / 256, 256, 0, stream>>>(Wa_w, wawh, wawl, WGN);
  split_kernel<<<(WGN + 255) / 256, 256, 0, stream>>>(Wi_w, wiwh, wiwl, WGN);
  split_kernel<<<(WFN + 255) / 256, 256, 0, stream>>>(Wf, wfh, wfl, WFN);

  // --- fused projection + gate + scan (one launch per path)
  dim3 gf(EXPN / 64, BATCH);  // (4, 32)
  fused_proj_scan_kernel<<<gf, 128, 0, stream>>>(xh0, xl0, wauh, waul, wiuh, wiul,
                                                 ba_u, bi_u, g_u, ch, cl, 0);
  fused_proj_scan_kernel<<<gf, 128, 0, stream>>>(xh1, xl1, wawh, wawl, wiwh, wiwl,
                                                 ba_w, bi_w, g_w, ch, cl, 1);

  // --- output GEMM
  dim3 go(M_TOTAL / 16, 2);
  out_gemm_kernel<<<go, 128, 0, stream>>>(ch, cl, wfh, wfl, bfv, out);
}
